// vt_model_12841952215611
// MI455X (gfx1250) — matrix operand order repaired and checked by reading, not compiled, not run on hardware
//
#include <hip/hip_runtime.h>
#include <stdint.h>

typedef __attribute__((ext_vector_type(16))) __bf16 v16bf;
typedef __attribute__((ext_vector_type(8)))  float  v8f;

#define IMG     224
#define GRID14  14
#define NPATCH  196
#define K_TOT   3840
#define KP      1920    // K_TOT/2: packed bf16x2 dwords per column
#define N_TOT   768
#define ROWS    64      // M rows per workgroup (12544 / 64 = 196 blocks)
#define KCHUNK  480     // 2 patch-rows: 2 * 16 px * 15 ch
#define NCHUNKS 8       // 3840 / 480
#define STEPS   15      // 480 / 32
#define SD      244     // LDS row stride in dwords (488 bf16 = 976 B, bank-padded)

union FragBF { uint32_t u[8]; v16bf v; };

__device__ __forceinline__ uint32_t pack_bf16(float lo, float hi) {
  // two fp32 -> packed bf16x2 (truncation)
  return (__float_as_uint(hi) & 0xFFFF0000u) | (__float_as_uint(lo) >> 16);
}

// ---------------------------------------------------------------------------
// Pre-pass: pack fp32 W (K,N) into bf16x2 dwords laid out column-major in the
// exact WMMA B-fragment order: Wp[n * KP + k/2] = (bf16 W[k][n], bf16 W[k+1][n]).
// A lane's 8 fragment dwords are then two 4-dword runs -> 2x global_load_b128.
// ---------------------------------------------------------------------------
__global__ __launch_bounds__(256) void pack_w_kernel(
    const float* __restrict__ Wm, uint32_t* __restrict__ Wp) {
  int idx = blockIdx.x * 256 + threadIdx.x;     // KP * N_TOT total
  int n  = idx / KP;
  int kp = idx - n * KP;
  float lo = Wm[(size_t)(2 * kp)     * N_TOT + n];
  float hi = Wm[(size_t)(2 * kp + 1) * N_TOT + n];
  Wp[idx] = pack_bf16(lo, hi);                  // coalesced writes
}

// ---------------------------------------------------------------------------
// Fused shifted-patch tokenization + GEMM (bf16 WMMA, fp32 accumulate)
// ---------------------------------------------------------------------------
__global__ __launch_bounds__(256) void spt_fused_gemm(
    const float*    __restrict__ images,   // (64,224,224,3)
    const uint32_t* __restrict__ Wp,       // packed bf16x2, [N][K/2]
    const float*    __restrict__ bias,     // (768)
    float*          __restrict__ out)      // (12544,768)
{
  __shared__ uint32_t lds[ROWS * SD];   // 62,464 bytes: A-chunk, bf16, padded

  const int tid  = threadIdx.x;
  const int lane = tid & 31;
  const int wave = tid >> 5;
  const int m0   = blockIdx.x * ROWS;
  const int ncol = blockIdx.y * 256 + wave * 32;   // this wave: cols [ncol, ncol+32)

  // ---- per-thread fill descriptors: 1024 pixel-pairs per chunk, 4 per thread
  int abase[4], wv[4], hb[4], lbase[4];
  #pragma unroll
  for (int i = 0; i < 4; ++i) {
    int p   = tid + 256 * i;       // pixel-pair index in [0,1024)
    int row = p >> 4;              // A row within tile (0..63)
    int t   = p & 15;
    int pyo = t >> 3;              // which of the 2 py rows in this chunk
    int q   = t & 7;               // px pair, px = 2q
    int r   = m0 + row;            // global A row
    int b   = r / NPATCH;
    int pa  = r - b * NPATCH;
    int gy  = pa / GRID14;
    int gx  = pa - gy * GRID14;
    int w   = gx * 16 + 2 * q;
    int h0  = gy * 16 + pyo;
    wv[i]    = w;
    hb[i]    = h0;
    abase[i] = b * (IMG * IMG * 3) + h0 * (IMG * 3) + w * 3;
    lbase[i] = row * SD + pyo * 120 + q * 15;   // dword offset in LDS
  }

  v8f acc[4][2];
  #pragma unroll
  for (int mi = 0; mi < 4; ++mi)
    #pragma unroll
    for (int ni = 0; ni < 2; ++ni)
      #pragma unroll
      for (int e = 0; e < 8; ++e) acc[mi][ni][e] = 0.0f;

  const int nlo   = lane & 15;
  const int ah    = (lane >> 4) * 4;   // A layout: upper lanes offset +8 bf16 (4 dw)
  // B base for this lane: column ncol+nlo, upper lanes offset +8 bf16 (+4 dwords)
  const uint32_t* wbase = Wp + (size_t)(ncol + nlo) * KP + ((lane >> 4) * 4);

  #pragma unroll 1
  for (int ch = 0; ch < NCHUNKS; ++ch) {
    __syncthreads();   // previous chunk's readers done

    // ---- stage A chunk (on-the-fly 5-way masked channel expansion -> bf16)
    #pragma unroll
    for (int i = 0; i < 4; ++i) {
      int h = hb[i] + ch * 2;
      const float* src = images + abase[i] + ch * (2 * IMG * 3);
      float v[6];
      #pragma unroll
      for (int j = 0; j < 6; ++j) v[j] = src[j];   // 2 pixels x 3 channels
      const bool hlt = h < (IMG - 8), hge = h >= 8;
      float a[30];
      #pragma unroll
      for (int px = 0; px < 2; ++px) {
        int w = wv[i] + px;
        const bool wlt = w < (IMG - 8), wge = w >= 8;
        float m1 = (hlt && wlt) ? 1.f : 0.f;   // offset (0,0)
        float m2 = (hge && wlt) ? 1.f : 0.f;   // offset (8,0)
        float m3 = (hlt && wge) ? 1.f : 0.f;   // offset (0,8)
        float m4 = (hge && wge) ? 1.f : 0.f;   // offset (8,8)
        #pragma unroll
        for (int c = 0; c < 3; ++c) {
          float x = v[px * 3 + c];
          a[px * 15 +  0 + c] = x;
          a[px * 15 +  3 + c] = x * m1;
          a[px * 15 +  6 + c] = x * m2;
          a[px * 15 +  9 + c] = x * m3;
          a[px * 15 + 12 + c] = x * m4;
        }
      }
      #pragma unroll
      for (int j = 0; j < 15; ++j)
        lds[lbase[i] + j] = pack_bf16(a[2 * j], a[2 * j + 1]);
    }
    __syncthreads();

    // ---- WMMA sweep over this K chunk
    const int kcd = (ch * KCHUNK) >> 1;   // dword offset into packed columns
    #pragma unroll 3
    for (int s = 0; s < STEPS; ++s) {
      // B fragments: 2 N-subtiles, 2x b128 each, straight from packed-bf16 W
      const uint32_t* wp = wbase + kcd + s * 16;
      FragBF fb0, fb1;
      uint4 q0 = *(const uint4*)(wp);
      uint4 q1 = *(const uint4*)(wp + 8);
      uint4 q2 = *(const uint4*)(wp + 16 * KP);
      uint4 q3 = *(const uint4*)(wp + 16 * KP + 8);
      fb0.u[0] = q0.x; fb0.u[1] = q0.y; fb0.u[2] = q0.z; fb0.u[3] = q0.w;
      fb0.u[4] = q1.x; fb0.u[5] = q1.y; fb0.u[6] = q1.z; fb0.u[7] = q1.w;
      fb1.u[0] = q2.x; fb1.u[1] = q2.y; fb1.u[2] = q2.z; fb1.u[3] = q2.w;
      fb1.u[4] = q3.x; fb1.u[5] = q3.y; fb1.u[6] = q3.z; fb1.u[7] = q3.w;

      #pragma unroll
      for (int mi = 0; mi < 4; ++mi) {
        FragBF fa;
        const uint32_t* ap = &lds[(mi * 16 + nlo) * SD + s * 16 + ah];
        uint4 lo = *(const uint4*)ap;        // K 0..7  (or 8..15 upper lanes)
        uint4 hi = *(const uint4*)(ap + 8);  // K 16..23 (or 24..31)
        fa.u[0] = lo.x; fa.u[1] = lo.y; fa.u[2] = lo.z; fa.u[3] = lo.w;
        fa.u[4] = hi.x; fa.u[5] = hi.y; fa.u[6] = hi.z; fa.u[7] = hi.w;
        acc[mi][0] = __builtin_amdgcn_wmma_f32_16x16x32_bf16(
            false, fa.v, false, fb0.v, (short)0, acc[mi][0], false, false);
        acc[mi][1] = __builtin_amdgcn_wmma_f32_16x16x32_bf16(
            false, fa.v, false, fb1.v, (short)0, acc[mi][1], false, false);
      }
    }
  }

  // ---- epilogue: bias + store (C/D layout: VGPR e -> M=e (lanes<16) / M=8+e)
  const float b0 = bias[ncol + nlo];
  const float b1 = bias[ncol + 16 + nlo];
  const int rbase = m0 + ((lane >> 4) << 3);
  #pragma unroll
  for (int mi = 0; mi < 4; ++mi)
    #pragma unroll
    for (int ni = 0; ni < 2; ++ni) {
      const float bb = ni ? b1 : b0;
      #pragma unroll
      for (int e = 0; e < 8; ++e) {
        int rr = rbase + mi * 16 + e;
        out[(size_t)rr * N_TOT + ncol + ni * 16 + nlo] = acc[mi][ni][e] + bb;
      }
    }
}

extern "C" void kernel_launch(void* const* d_in, const int* in_sizes, int n_in,
                              void* d_out, int out_size, void* d_ws, size_t ws_size,
                              hipStream_t stream) {
  (void)in_sizes; (void)n_in; (void)out_size; (void)ws_size;
  const float* images = (const float*)d_in[0];  // (64,224,224,3)
  const float* Wm     = (const float*)d_in[1];  // (3840,768)
  const float* bias   = (const float*)d_in[2];  // (768)
  float* out          = (float*)d_out;          // (12544,768)
  uint32_t* Wp        = (uint32_t*)d_ws;        // 5,898,240 bytes packed bf16 W

  // 1) pack W -> bf16x2 fragment layout (KP*N_TOT = 1,474,560 dwords)
  pack_w_kernel<<<dim3((KP * N_TOT) / 256), dim3(256), 0, stream>>>(Wm, Wp);

  // 2) fused tokenization + GEMM
  dim3 grid(196, 3, 1);   // 196 M-blocks of 64 rows x 3 N-blocks of 256 cols
  dim3 block(256, 1, 1);  // 8 wave32
  spt_fused_gemm<<<grid, block, 0, stream>>>(images, Wp, bias, out);
}
